// TransformerEncoderLayer_12979391169220
// MI455X (gfx1250) — matrix operand order repaired and checked by reading, not compiled, not run on hardware
//
#include <hip/hip_runtime.h>
#include <hip/hip_bf16.h>

typedef __attribute__((ext_vector_type(16))) _Float16 v16h;
typedef __attribute__((ext_vector_type(8)))  float    v8f;
typedef __attribute__((ext_vector_type(8)))  unsigned int v8u;

static __device__ __forceinline__ v8f wmma_f16(v8u a, v8u b, v8f c) {
    return __builtin_amdgcn_wmma_f32_16x16x32_f16(
        /*neg_a=*/false, __builtin_bit_cast(v16h, a),
        /*neg_b=*/false, __builtin_bit_cast(v16h, b),
        /*c_mod=*/(short)0, c, /*reuse_a=*/false, /*reuse_b=*/false);
}

// ---------------------------------------------------------------------------
// Weight convert + transpose: W fp32 [K][N]  ->  WT f16 [N][K]
// ---------------------------------------------------------------------------
__global__ __launch_bounds__(256) void convert_transpose_kernel(
    const float* __restrict__ W, _Float16* __restrict__ WT, int K, int N)
{
    int idx = blockIdx.x * 256 + threadIdx.x;
    if (idx >= K * N) return;
    int kk = idx / N, nn = idx % N;
    WT[(size_t)nn * K + kk] = (_Float16)W[idx];
}

// ---------------------------------------------------------------------------
// LayerNorm over last dim D (D <= 768), optional residual add.
// Emits fp32 and/or f16 outputs.
// ---------------------------------------------------------------------------
__global__ __launch_bounds__(256) void layernorm_kernel(
    const float* __restrict__ x, const float* __restrict__ res,
    const float* __restrict__ g, const float* __restrict__ bta,
    float* __restrict__ of, _Float16* __restrict__ oh, int D)
{
    __shared__ float s1[256];
    __shared__ float s2[256];
    int row = blockIdx.x, tid = threadIdx.x;
    const float* xr = x + (size_t)row * D;
    const float* rr = res ? res + (size_t)row * D : nullptr;
    float lv[3];
    int cnt = 0;
    float a = 0.0f, b = 0.0f;
    for (int i = tid; i < D; i += 256) {
        float vv = xr[i] + (rr ? rr[i] : 0.0f);
        lv[cnt++] = vv;
        a += vv; b += vv * vv;
    }
    s1[tid] = a; s2[tid] = b;
    __syncthreads();
    for (int st = 128; st > 0; st >>= 1) {
        if (tid < st) { s1[tid] += s1[tid + st]; s2[tid] += s2[tid + st]; }
        __syncthreads();
    }
    float mean = s1[0] / (float)D;
    float var  = s2[0] / (float)D - mean * mean;
    float rstd = rsqrtf(var + 1e-5f);
    cnt = 0;
    for (int i = tid; i < D; i += 256) {
        float vv = (lv[cnt++] - mean) * rstd * g[i] + bta[i];
        size_t o = (size_t)row * D + i;
        if (of) of[o] = vv;
        if (oh) oh[o] = (_Float16)vv;
    }
}

// ---------------------------------------------------------------------------
// WMMA GEMM: C[M,N] = act(A[M,K] @ B[K,N] + bias) + resid
//   A: f16 row-major [M][K]; BT: f16 [N][K] (B transposed -> K-contiguous).
//   256 threads = 8 waves arranged 2x4; each wave computes a 32x64 tile
//   (2 A-frags x 4 B-frags = 8 WMMAs / 32-k chunk); block tile = 64x256.
//   act: 0 = none, 1 = exact GELU. Outputs fp32 and/or f16.
// ---------------------------------------------------------------------------
__global__ __launch_bounds__(256) void gemm_f16_kernel(
    const _Float16* __restrict__ A, const _Float16* __restrict__ BT,
    const float* __restrict__ bias, const float* __restrict__ resid,
    float* __restrict__ Cf, _Float16* __restrict__ Ch,
    int M, int N, int K, int act)
{
    int tid  = threadIdx.x;
    int wave = tid >> 5, lane = tid & 31;
    int wr = wave >> 2, wc = wave & 3;
    int row0 = blockIdx.y * 64 + wr * 32;
    int col0 = blockIdx.x * 256 + wc * 64;
    int m = lane & 15, hi = lane >> 4;

    const unsigned int* Ar0 = (const unsigned int*)(A + (size_t)(row0 + m) * K);
    const unsigned int* Ar1 = (const unsigned int*)(A + (size_t)(row0 + 16 + m) * K);
    const unsigned int* Br0 = (const unsigned int*)(BT + (size_t)(col0 + m) * K);
    const unsigned int* Br1 = (const unsigned int*)(BT + (size_t)(col0 + 16 + m) * K);
    const unsigned int* Br2 = (const unsigned int*)(BT + (size_t)(col0 + 32 + m) * K);
    const unsigned int* Br3 = (const unsigned int*)(BT + (size_t)(col0 + 48 + m) * K);
    const unsigned int* Brs[4] = { Br0, Br1, Br2, Br3 };

    v8f zero = {};
    v8f acc[2][4];
#pragma unroll
    for (int g = 0; g < 2; ++g)
#pragma unroll
        for (int c = 0; c < 4; ++c) acc[g][c] = zero;

    for (int k0 = 0; k0 < K; k0 += 32) {
        v8u a0, a1;
        int ab = k0 + hi * 16;
#pragma unroll
        for (int vi = 0; vi < 4; ++vi) {
            a0[vi]     = Ar0[(ab + 2 * vi) >> 1];
            a0[4 + vi] = Ar0[(ab + 8 + 2 * vi) >> 1];
            a1[vi]     = Ar1[(ab + 2 * vi) >> 1];
            a1[4 + vi] = Ar1[(ab + 8 + 2 * vi) >> 1];
        }
        int bb = k0 + hi * 16;
#pragma unroll
        for (int c = 0; c < 4; ++c) {
            v8u bu;
#pragma unroll
            for (int vi = 0; vi < 8; ++vi) bu[vi] = Brs[c][(bb + 2 * vi) >> 1];
            acc[0][c] = wmma_f16(a0, bu, acc[0][c]);
            acc[1][c] = wmma_f16(a1, bu, acc[1][c]);
        }
        if (k0 + 64 < K) {  // hide HBM/L2 latency of the next k-chunk
            __builtin_prefetch((const char*)&Ar0[(k0 + 64) >> 1], 0, 0);
            __builtin_prefetch((const char*)&Ar1[(k0 + 64) >> 1], 0, 0);
            __builtin_prefetch((const char*)&Br0[(k0 + 64) >> 1], 0, 0);
        }
    }

#pragma unroll
    for (int g = 0; g < 2; ++g) {
#pragma unroll
        for (int c = 0; c < 4; ++c) {
            int col = col0 + c * 16 + m;
            float bv = bias ? bias[col] : 0.0f;
#pragma unroll
            for (int r = 0; r < 8; ++r) {
                int row = row0 + g * 16 + r + hi * 8;
                float v = acc[g][c][r] + bv;
                if (act == 1) v = 0.5f * v * (1.0f + erff(v * 0.70710678118654752f));
                size_t o = (size_t)row * N + col;
                if (resid) v += resid[o];
                if (Cf) Cf[o] = v;
                if (Ch) Ch[o] = (_Float16)v;
            }
        }
    }
}

// ---------------------------------------------------------------------------
// Flash cross-attention: ctx = softmax(Q K^T / 8) V per (b, head).
// Block = 4 waves sharing one (b, h); each wave owns 16 query rows.
// Per 32-key step: V block staged TRANSPOSED into LDS cooperatively
// (coalesced global reads, key-contiguous fragment reads), 4 score WMMAs,
// online softmax, P repacked via LDS, 4 ctx WMMAs.
// ---------------------------------------------------------------------------
__global__ __launch_bounds__(128) void flash_attn_kernel(
    const _Float16* __restrict__ q, const _Float16* __restrict__ k,
    const _Float16* __restrict__ v, _Float16* __restrict__ ctx)
{
    __shared__ _Float16 vt[64 * 32];      // V^T block: [d(64)][key(32)]
    __shared__ _Float16 pb[4][16 * 32];   // per-wave P tile
    const int S = 2048, D = 512, HD = 64;
    int tid = threadIdx.x;
    int w = tid >> 5, lane = tid & 31;
    int m = lane & 15, hi = lane >> 4;
    int b = blockIdx.z, h = blockIdx.y;
    int qbase = (blockIdx.x * 4 + w) * 16;

    // Q fragments (A-matrix 16x32 layout), d-chunks [0,32) and [32,64)
    const unsigned int* qr = (const unsigned int*)(q + (size_t)(b * S + qbase + m) * D + h * HD);
    v8u a0, a1;
    {
        int ab = hi * 16;
#pragma unroll
        for (int vi = 0; vi < 4; ++vi) {
            a0[vi]     = qr[(ab + 2 * vi) >> 1];
            a0[4 + vi] = qr[(ab + 8 + 2 * vi) >> 1];
            a1[vi]     = qr[(32 + ab + 2 * vi) >> 1];
            a1[4 + vi] = qr[(32 + ab + 8 + 2 * vi) >> 1];
        }
    }

    float mrow[8], lrow[8];
    v8f zero = {};
    v8f oacc[4];
#pragma unroll
    for (int c = 0; c < 4; ++c) oacc[c] = zero;
#pragma unroll
    for (int r = 0; r < 8; ++r) { mrow[r] = -1e30f; lrow[r] = 0.0f; }

    // cooperative staging indices: thread -> (key, 16-wide d group)
    int skey = tid >> 2;
    int sdg  = (tid & 3) * 16;

    for (int kb0 = 0; kb0 < S; kb0 += 32) {
        // ---- stage V^T block into LDS (coalesced 32B reads per thread)
        {
            const unsigned int* src =
                (const unsigned int*)(v + (size_t)(b * S + kb0 + skey) * D + h * HD + sdg);
            union { unsigned int u[8]; _Float16 hl[16]; } tb;
#pragma unroll
            for (int i = 0; i < 8; ++i) tb.u[i] = src[i];
#pragma unroll
            for (int i = 0; i < 16; ++i) vt[(sdg + i) * 32 + skey] = tb.hl[i];
            if (kb0 + 32 < S) {  // prefetch next K/V key-block
                __builtin_prefetch((const char*)(v + (size_t)(b * S + kb0 + 32 + skey) * D + h * HD + sdg), 0, 0);
                __builtin_prefetch((const char*)(k + (size_t)(b * S + kb0 + 32 + skey) * D + h * HD + sdg), 0, 0);
            }
        }
        __syncthreads();

        // ---- scores: two 16x16 tiles over keys [kb0,kb0+16) and [kb0+16,kb0+32)
        v8f s0 = zero, s1 = zero;
        {
            const unsigned int* kr0 = (const unsigned int*)(k + (size_t)(b * S + kb0 + m) * D + h * HD);
            const unsigned int* kr1 = (const unsigned int*)(k + (size_t)(b * S + kb0 + 16 + m) * D + h * HD);
            int bb = hi * 16;
            v8u bu;
#pragma unroll
            for (int vi = 0; vi < 8; ++vi) bu[vi] = kr0[(bb + 2 * vi) >> 1];
            s0 = wmma_f16(a0, bu, s0);
#pragma unroll
            for (int vi = 0; vi < 8; ++vi) bu[vi] = kr0[(32 + bb + 2 * vi) >> 1];
            s0 = wmma_f16(a1, bu, s0);
#pragma unroll
            for (int vi = 0; vi < 8; ++vi) bu[vi] = kr1[(bb + 2 * vi) >> 1];
            s1 = wmma_f16(a0, bu, s1);
#pragma unroll
            for (int vi = 0; vi < 8; ++vi) bu[vi] = kr1[(32 + bb + 2 * vi) >> 1];
            s1 = wmma_f16(a1, bu, s1);
        }

        // ---- online softmax update; write P tile (f16) to this wave's LDS slice
#pragma unroll
        for (int r = 0; r < 8; ++r) {
            float x0 = s0[r] * 0.125f, x1 = s1[r] * 0.125f;
            float mx = fmaxf(x0, x1);
            for (int d = 1; d < 16; d <<= 1) mx = fmaxf(mx, __shfl_xor(mx, d, 32));
            float mn = fmaxf(mrow[r], mx);
            float corr = __expf(mrow[r] - mn);
            float p0 = __expf(x0 - mn), p1 = __expf(x1 - mn);
            float ps = p0 + p1;
            for (int d = 1; d < 16; d <<= 1) ps += __shfl_xor(ps, d, 32);
            lrow[r] = lrow[r] * corr + ps;
            mrow[r] = mn;
            oacc[0][r] *= corr; oacc[1][r] *= corr;
            oacc[2][r] *= corr; oacc[3][r] *= corr;
            int row = r + hi * 8;
            pb[w][row * 32 + m]      = (_Float16)p0;
            pb[w][row * 32 + 16 + m] = (_Float16)p1;
        }

        // ---- re-pack P as A-fragment (16x32) from LDS
        v8u pf;
        {
            const _Float16* pbase = &pb[w][0];
#pragma unroll
            for (int vi = 0; vi < 4; ++vi) {
                unsigned int t0, t1;
                __builtin_memcpy(&t0, pbase + m * 32 + hi * 16 + 2 * vi, 4);
                __builtin_memcpy(&t1, pbase + m * 32 + hi * 16 + 8 + 2 * vi, 4);
                pf[vi] = t0; pf[4 + vi] = t1;
            }
        }

        // ---- ctx += P(16x32) @ V(32x64): B-frags from V^T in LDS (key-contiguous)
#pragma unroll
        for (int c = 0; c < 4; ++c) {
            const _Float16* vrow = &vt[(c * 16 + m) * 32 + hi * 16];
            v8u bu;
#pragma unroll
            for (int vi = 0; vi < 8; ++vi) {
                unsigned int t;
                __builtin_memcpy(&t, vrow + 2 * vi, 4);
                bu[vi] = t;
            }
            oacc[c] = wmma_f16(pf, bu, oacc[c]);
        }
        __syncthreads();   // all waves done with vt before next staging
    }

    // ---- normalize and store ctx (f16)
    size_t obase = (size_t)(b * S + qbase) * D + h * HD;
#pragma unroll
    for (int r = 0; r < 8; ++r) {
        float inv = 1.0f / lrow[r];
        int row = r + hi * 8;
#pragma unroll
        for (int c = 0; c < 4; ++c)
            ctx[obase + (size_t)row * D + c * 16 + m] = (_Float16)(oacc[c][r] * inv);
    }
}

// ---------------------------------------------------------------------------
// Degenerate MHA (batch_first=False quirk): attention over L=4, per (n, h, l).
// qkv fp32 [L*N][1536]; writes ctx2 f16 [L*N][512].
// ---------------------------------------------------------------------------
__global__ __launch_bounds__(256) void mini_attn_kernel(
    const float* __restrict__ qkv, _Float16* __restrict__ ctx2)
{
    const int Nn = 2048, D3 = 1536, D = 512, HD = 64;
    int idx = blockIdx.x * 256 + threadIdx.x;
    int l = idx & 3;
    int h = (idx >> 2) & 7;
    int n = idx >> 5;
    const float* qrow = qkv + (size_t)(l * Nn + n) * D3 + h * HD;
    float s[4];
#pragma unroll
    for (int mm = 0; mm < 4; ++mm) {
        const float* krow = qkv + (size_t)(mm * Nn + n) * D3 + D + h * HD;
        float acc = 0.0f;
        for (int d = 0; d < HD; ++d) acc += qrow[d] * krow[d];
        s[mm] = acc * 0.125f;
    }
    float mx = fmaxf(fmaxf(s[0], s[1]), fmaxf(s[2], s[3]));
    float e[4], se = 0.0f;
#pragma unroll
    for (int mm = 0; mm < 4; ++mm) { e[mm] = __expf(s[mm] - mx); se += e[mm]; }
    float inv = 1.0f / se;
    _Float16* out = ctx2 + (size_t)(l * Nn + n) * D + h * HD;
    for (int d = 0; d < HD; ++d) {
        float acc = 0.0f;
#pragma unroll
        for (int mm = 0; mm < 4; ++mm)
            acc += e[mm] * qkv[(size_t)(mm * Nn + n) * D3 + 2 * D + h * HD + d];
        out[d] = (_Float16)(acc * inv);
    }
}

// ---------------------------------------------------------------------------
extern "C" void kernel_launch(void* const* d_in, const int* in_sizes, int n_in,
                              void* d_out, int out_size, void* d_ws, size_t ws_size,
                              hipStream_t stream) {
    const int B = 4, S = 2048, Dq = 512, Dk = 768, Dff = 2048;
    const int BS = B * S;  // 8192

    const float* x        = (const float*)d_in[0];
    const float* y        = (const float*)d_in[1];
    const float* Wq       = (const float*)d_in[2];
    const float* Wk       = (const float*)d_in[3];
    const float* Wv       = (const float*)d_in[4];
    const float* Wo       = (const float*)d_in[5];
    const float* bo       = (const float*)d_in[6];
    const float* mha_w_in = (const float*)d_in[7];
    const float* mha_b_in = (const float*)d_in[8];
    const float* mha_wo   = (const float*)d_in[9];
    const float* mha_bo   = (const float*)d_in[10];
    const float* ln1_g = (const float*)d_in[11]; const float* ln1_b = (const float*)d_in[12];
    const float* ln2_g = (const float*)d_in[13]; const float* ln2_b = (const float*)d_in[14];
    const float* ln3_g = (const float*)d_in[15]; const float* ln3_b = (const float*)d_in[16];
    const float* ln4_g = (const float*)d_in[17]; const float* ln4_b = (const float*)d_in[18];
    const float* mlp_w1 = (const float*)d_in[19]; const float* mlp_b1 = (const float*)d_in[20];
    const float* mlp_w2 = (const float*)d_in[21]; const float* mlp_b2 = (const float*)d_in[22];
    float* out = (float*)d_out;

    // ---- carve workspace
    char* wsp = (char*)d_ws;
    size_t off = 0;
    auto alloc = [&](size_t bytes) -> char* {
        char* p = wsp + off;
        off = (off + bytes + 255) & ~(size_t)255;
        return p;
    };
    float*    xn_f32   = (float*)alloc((size_t)BS * Dq * 4);
    float*    attn_f32 = (float*)alloc((size_t)BS * Dq * 4);   // ca, then sa
    float*    h_f32    = (float*)alloc((size_t)BS * Dq * 4);
    float*    h2_f32   = (float*)alloc((size_t)BS * Dq * 4);
    float*    qkv_f32  = (float*)alloc((size_t)BS * 3 * Dq * 4);
    _Float16* bufA_h   = (_Float16*)alloc((size_t)BS * Dq * 2); // xn_h, then ctx2_h
    _Float16* yn_h     = (_Float16*)alloc((size_t)BS * Dk * 2);
    _Float16* big_h    = (_Float16*)alloc((size_t)BS * Dff * 2); // q|k|v|ctx, then mlp1
    _Float16* h_h      = (_Float16*)alloc((size_t)BS * Dq * 2);
    _Float16* h2_h     = (_Float16*)alloc((size_t)BS * Dq * 2);
    _Float16* WqT  = (_Float16*)alloc((size_t)Dq * Dq * 2);
    _Float16* WkT  = (_Float16*)alloc((size_t)Dq * Dk * 2);
    _Float16* WvT  = (_Float16*)alloc((size_t)Dq * Dk * 2);
    _Float16* WoT  = (_Float16*)alloc((size_t)Dq * Dq * 2);
    _Float16* WinT = (_Float16*)alloc((size_t)3 * Dq * Dq * 2);
    _Float16* WmoT = (_Float16*)alloc((size_t)Dq * Dq * 2);
    _Float16* W1T  = (_Float16*)alloc((size_t)Dff * Dq * 2);
    _Float16* W2T  = (_Float16*)alloc((size_t)Dq * Dff * 2);

    _Float16* xn_h   = bufA_h;
    _Float16* ctx2_h = bufA_h;
    _Float16* q_h    = big_h;
    _Float16* k_h    = big_h + (size_t)BS * Dq;
    _Float16* v_h    = big_h + (size_t)2 * BS * Dq;
    _Float16* ctx_h  = big_h + (size_t)3 * BS * Dq;
    _Float16* mlp1_h = big_h;

    auto cgrid = [](int n) { return dim3((unsigned)((n + 255) / 256)); };

    // ---- 1) weights -> f16 transposed [N][K]
    convert_transpose_kernel<<<cgrid(Dq * Dq), 256, 0, stream>>>(Wq, WqT, Dq, Dq);
    convert_transpose_kernel<<<cgrid(Dk * Dq), 256, 0, stream>>>(Wk, WkT, Dk, Dq);
    convert_transpose_kernel<<<cgrid(Dk * Dq), 256, 0, stream>>>(Wv, WvT, Dk, Dq);
    convert_transpose_kernel<<<cgrid(Dq * Dq), 256, 0, stream>>>(Wo, WoT, Dq, Dq);
    convert_transpose_kernel<<<cgrid(Dq * 3 * Dq), 256, 0, stream>>>(mha_w_in, WinT, Dq, 3 * Dq);
    convert_transpose_kernel<<<cgrid(Dq * Dq), 256, 0, stream>>>(mha_wo, WmoT, Dq, Dq);
    convert_transpose_kernel<<<cgrid(Dq * Dff), 256, 0, stream>>>(mlp_w1, W1T, Dq, Dff);
    convert_transpose_kernel<<<cgrid(Dff * Dq), 256, 0, stream>>>(mlp_w2, W2T, Dff, Dq);

    // ---- 2) LN1(x) -> xn (fp32 + f16);  LN2(y) -> yn (f16)
    layernorm_kernel<<<BS, 256, 0, stream>>>(x, nullptr, ln1_g, ln1_b, xn_f32, xn_h, Dq);
    layernorm_kernel<<<BS, 256, 0, stream>>>(y, nullptr, ln2_g, ln2_b, nullptr, yn_h, Dk);

    // ---- 3) q/k/v projections (f16 out)
    gemm_f16_kernel<<<dim3(Dq / 256, BS / 64), 256, 0, stream>>>(
        xn_h, WqT, nullptr, nullptr, nullptr, q_h, BS, Dq, Dq, 0);
    gemm_f16_kernel<<<dim3(Dq / 256, BS / 64), 256, 0, stream>>>(
        yn_h, WkT, nullptr, nullptr, nullptr, k_h, BS, Dq, Dk, 0);
    gemm_f16_kernel<<<dim3(Dq / 256, BS / 64), 256, 0, stream>>>(
        yn_h, WvT, nullptr, nullptr, nullptr, v_h, BS, Dq, Dk, 0);

    // ---- 4) flash cross-attention
    flash_attn_kernel<<<dim3(S / 16 / 4, 8, B), 128, 0, stream>>>(q_h, k_h, v_h, ctx_h);

    // ---- 5) out-projection -> ca (fp32)
    gemm_f16_kernel<<<dim3(Dq / 256, BS / 64), 256, 0, stream>>>(
        ctx_h, WoT, bo, nullptr, attn_f32, nullptr, BS, Dq, Dq, 0);

    // ---- 6) h = LN3(ca + xn)
    layernorm_kernel<<<BS, 256, 0, stream>>>(attn_f32, xn_f32, ln3_g, ln3_b, h_f32, h_h, Dq);

    // ---- 7) packed qkv in-projection (fp32 out)
    gemm_f16_kernel<<<dim3(3 * Dq / 256, BS / 64), 256, 0, stream>>>(
        h_h, WinT, mha_b_in, nullptr, qkv_f32, nullptr, BS, 3 * Dq, Dq, 0);

    // ---- 8) degenerate L=4 self-attention -> ctx2 (f16)
    mini_attn_kernel<<<dim3((B * S * 8) / 256), 256, 0, stream>>>(qkv_f32, ctx2_h);

    // ---- 9) MHA out-projection -> sa (fp32)
    gemm_f16_kernel<<<dim3(Dq / 256, BS / 64), 256, 0, stream>>>(
        ctx2_h, WmoT, mha_bo, nullptr, attn_f32, nullptr, BS, Dq, Dq, 0);

    // ---- 10) h2 = LN4(sa + h)
    layernorm_kernel<<<BS, 256, 0, stream>>>(attn_f32, h_f32, ln4_g, ln4_b, h2_f32, h2_h, Dq);

    // ---- 11) MLP: gelu(h2 @ w1 + b1) -> f16, then @ w2 + b2 + h2 -> out
    gemm_f16_kernel<<<dim3(Dff / 256, BS / 64), 256, 0, stream>>>(
        h2_h, W1T, mlp_b1, nullptr, nullptr, mlp1_h, BS, Dff, Dq, 1);
    gemm_f16_kernel<<<dim3(Dq / 256, BS / 64), 256, 0, stream>>>(
        mlp1_h, W2T, mlp_b2, h2_f32, out, nullptr, BS, Dq, Dff, 0);
}
